// EnhancedAMSR2Loss_47416438948373
// MI455X (gfx1250) — matrix operand order repaired and checked by reading, not compiled, not run on hardware
//
#include <hip/hip_runtime.h>
#include <math.h>

// ---------------------------------------------------------------------------
// EnhancedAMSR2Loss for MI455X (gfx1250, wave32).
//  Pass 1: fused elementwise stats reduction (L1, grads, per-image mean/var,
//          range penalty, PSNR mse on uint8-quantized images).
//  Pass 2: SSIM via separable 11x11 Gaussian. Horizontal conv is done with
//          v_wmma_f32_16x16x32_f16 (banded weight matrix as B operand);
//          vertical conv in exact f32 through per-wave LDS scratch.
//  Pass 3: tiny finalize kernel combining the 25 double accumulators.
// Memory-bound workload: 128 MiB of inputs (fits in 192MB L2), HBM floor
// ~5.6us @ 23.3 TB/s; WMMA keeps the conv off the VALU for ~free.
// ---------------------------------------------------------------------------

typedef __attribute__((ext_vector_type(16))) _Float16 v16h;
typedef __attribute__((ext_vector_type(8)))  float    v8f;

#define BATCH 4
#define Hd 2048
#define Wd 2048
#define IMG (Hd * Wd)          // 4,194,304 = 2^22
#define OUTD 2038              // 2048 - 10 (VALID 11x11)
#define C1v 6.5025f            // (0.01*255)^2
#define C2v 58.5225f           // (0.03*255)^2

// ws layout (doubles):
//  [0..3]  sumP[b]   [4..7]  sumT[b]   [8..11] sumP2[b]  [12..15] sumT2[b]
//  [16..19] mse8[b]  [20] sumAbs  [21] gxSum  [22] gySum  [23] rangeSum
//  [24] ssimSum
#define WS_DOUBLES 25

__device__ __forceinline__ float q255(float x) {
  // floor(clip(clip((clip(x,-1,1)+1)*0.5,0,1)*255,0,255))
  x = fminf(fmaxf(x, -1.f), 1.f);
  float y = (x + 1.f) * 0.5f;
  y = fminf(fmaxf(y, 0.f), 1.f);
  return floorf(fminf(y * 255.f, 255.f));
}

__device__ __forceinline__ float wred(float v) {
#pragma unroll
  for (int o = 16; o > 0; o >>= 1) v += __shfl_down(v, o, 32);
  return v;
}

// ---------------------------------------------------------------------------
// Pass 1: fused stats. One block = 1024 consecutive pixels (never straddles
// an image since IMG % 1024 == 0). float4 vector loads; down-row loads for gx
// hit L2 on re-use.
// ---------------------------------------------------------------------------
__global__ void __launch_bounds__(256)
stats_kernel(const float* __restrict__ p, const float* __restrict__ t,
             double* __restrict__ ws) {
  const int tid = threadIdx.x;
  const long long idx = ((long long)blockIdx.x * 256 + tid) * 4;
  const int b = (int)(idx >> 22);
  const int h = (int)((idx >> 11) & 2047);
  const int c = (int)(idx & 2047);

  float4 p4 = *(const float4*)(p + idx);
  float4 t4 = *(const float4*)(t + idx);
  float4 pD = p4, tD = t4;
  const bool hasD = (h < Hd - 1);
  if (hasD) {
    pD = *(const float4*)(p + idx + Wd);
    tD = *(const float4*)(t + idx + Wd);
  }
  float pR = 0.f, tR = 0.f;
  if (c + 4 < Wd) { pR = p[idx + 4]; tR = t[idx + 4]; }

  const float pe[5] = {p4.x, p4.y, p4.z, p4.w, pR};
  const float te[5] = {t4.x, t4.y, t4.z, t4.w, tR};
  const float pde[4] = {pD.x, pD.y, pD.z, pD.w};
  const float tde[4] = {tD.x, tD.y, tD.z, tD.w};

  float sP = 0, sT = 0, sP2 = 0, sT2 = 0, sMse = 0;
  float sAbs = 0, sGx = 0, sGy = 0, sRng = 0;
#pragma unroll
  for (int e = 0; e < 4; e++) {
    const float pv = pe[e], tv = te[e];
    sAbs += fabsf(pv - tv);
    sP += pv; sT += tv; sP2 += pv * pv; sT2 += tv * tv;
    sRng += fmaxf(fabsf(pv) - 1.f, 0.f);
    const float qd = q255(pv) - q255(tv);
    sMse += qd * qd;
    if (hasD) sGx += fabsf((pv - pde[e]) - (tv - tde[e]));
    if (c + e < Wd - 1) sGy += fabsf((pv - pe[e + 1]) - (tv - te[e + 1]));
  }

  float vals[9] = {sP, sT, sP2, sT2, sMse, sAbs, sGx, sGy, sRng};
  __shared__ float sh[9][8];
  const int lane = tid & 31, wid = tid >> 5;
#pragma unroll
  for (int q = 0; q < 9; q++) {
    const float r = wred(vals[q]);
    if (lane == 0) sh[q][wid] = r;
  }
  __syncthreads();
  if (wid == 0) {
#pragma unroll
    for (int q = 0; q < 9; q++) {
      float x = (lane < 8) ? sh[q][lane] : 0.f;
      x = wred(x);
      if (lane == 0) {
        const int slot = (q < 5) ? (q * 4 + b) : (20 + (q - 5));
        atomicAdd(ws + slot, (double)x);
      }
    }
  }
}

// ---------------------------------------------------------------------------
// Pass 2: SSIM. One wave per 16x16 output tile; 8 waves per block.
// Horizontal Gaussian conv as WMMA (A = 16x32 f16 window rows, B = banded
// 32x16 weight matrix), two WMMAs per conv input (window rows 0-15, 16-31),
// five conv inputs {p8, t8, p8^2, t8^2, p8*t8}. Vertical 11-tap conv done in
// f32 through per-wave LDS scratch hc[26][17].
// ---------------------------------------------------------------------------
__global__ void __launch_bounds__(256)
ssim_kernel(const float* __restrict__ pred, const float* __restrict__ targ,
            double* __restrict__ ws) {
  __shared__ float hc[8][26][17];  // per-wave, padded stride vs bank conflicts

  const int tid = threadIdx.x, lane = tid & 31, w = tid >> 5;
  const int tileId = blockIdx.x * 8 + w;
  const int b = tileId >> 14;            // /(128*128)
  const int rem = tileId & 16383;
  const int ty = rem >> 7, tx = rem & 127;
  const float* p = pred + (size_t)b * IMG;
  const float* t = targ + (size_t)b * IMG;

  // 11-tap Gaussian(sigma=1.5), normalized (cv2.getGaussianKernel(11,1.5))
  float gw[11];
  float gs = 0.f;
#pragma unroll
  for (int d = 0; d < 11; d++) {
    const float z = (float)(d - 5);
    gw[d] = expf(-z * z / 4.5f);
    gs += gw[d];
  }
#pragma unroll
  for (int d = 0; d < 11; d++) gw[d] /= gs;

  // B fragment: dense 16-bit B (32x16), same K order as A:
  // lanes 0-15 hold K in {0..7, 16..23}, lanes 16-31 hold K in {8..15, 24..31};
  // v16h element e -> K = kB0 + (e<8 ? e : e+8). B[k][n] = g[k-n] for k-n in [0,10], else 0
  // (zero for k>=26 covers the K-padding of A automatically).
  const int n = lane & 15;
  const int kB0 = (lane < 16) ? 0 : 8;
  v16h Bf;
#pragma unroll
  for (int e = 0; e < 16; e++) {
    const int d = (kB0 + (e < 8 ? e : e + 8)) - n;
    Bf[e] = (d >= 0 && d <= 10) ? (_Float16)gw[d] : (_Float16)0.0f;
  }

  // A-fragment source data. 16-bit A (16x32): lane<16 owns row M=lane,
  // K in {0..7, 16..23}; lane>=16 owns row M=lane-16, K in {8..15, 24..31};
  // v16h element e -> K = kb0 + (e<8 ? e : e+8). Both 8-wide column groups are
  // contiguous & 8-float aligned -> two float4 loads each. Clamped loads only
  // ever feed masked outputs / zero B entries.
  const int m = lane & 15;
  const int kb0 = (lane < 16) ? 0 : 8;
  const int cb = tx * 16;
  const int rtop = min(ty * 16 + m, Hd - 1);
  const int rbot = min(ty * 16 + 16 + m, Hd - 1);
  const int c0 = min(cb + kb0, Wd - 8);
  const int c1 = min(cb + kb0 + 16, Wd - 8);

  float qa[2][16], qb[2][16];
#pragma unroll
  for (int rh = 0; rh < 2; rh++) {
    const int r = rh ? rbot : rtop;
    const float* pr = p + (size_t)r * Wd;
    const float* tr = t + (size_t)r * Wd;
    const float4 pA = *(const float4*)(pr + c0), pB = *(const float4*)(pr + c0 + 4);
    const float4 pC = *(const float4*)(pr + c1), pD = *(const float4*)(pr + c1 + 4);
    const float4 tA = *(const float4*)(tr + c0), tB = *(const float4*)(tr + c0 + 4);
    const float4 tC = *(const float4*)(tr + c1), tD = *(const float4*)(tr + c1 + 4);
    const float pv[16] = {pA.x, pA.y, pA.z, pA.w, pB.x, pB.y, pB.z, pB.w,
                          pC.x, pC.y, pC.z, pC.w, pD.x, pD.y, pD.z, pD.w};
    const float tv[16] = {tA.x, tA.y, tA.z, tA.w, tB.x, tB.y, tB.z, tB.w,
                          tC.x, tC.y, tC.z, tC.w, tD.x, tD.y, tD.z, tD.w};
#pragma unroll
    for (int e = 0; e < 16; e++) {
      qa[rh][e] = q255(pv[e]);
      qb[rh][e] = q255(tv[e]);
    }
  }

  const int Mo = (lane < 16) ? 0 : 8;     // C-layout row offset
  const int mbase = Mo;                   // this lane's 8 output rows
  float res[5][8];

#pragma unroll
  for (int s = 0; s < 5; s++) {
    v16h Atop, Abot;
#pragma unroll
    for (int rh = 0; rh < 2; rh++) {
#pragma unroll
      for (int e = 0; e < 16; e++) {
        const float av = qa[rh][e], bv = qb[rh][e];
        const float val = (s == 0) ? av
                        : (s == 1) ? bv
                        : (s == 2) ? av * av
                        : (s == 3) ? bv * bv
                        : av * bv;
        if (rh == 0) Atop[e] = (_Float16)val;
        else         Abot[e] = (_Float16)val;
      }
    }
    v8f czero = {};
    // Hc[win_row][n] = sum_d g[d] * q[win_row][tx*16+n+d]
    v8f ctop = __builtin_amdgcn_wmma_f32_16x16x32_f16(
        false, Atop, false, Bf, (short)0, czero, false, false);
    v8f cbot = __builtin_amdgcn_wmma_f32_16x16x32_f16(
        false, Abot, false, Bf, (short)0, czero, false, false);

    // C layout: lane l, VGPR v -> (M = v + Mo, N = l%16). Stage window rows
    // 0..25 into LDS (rows 26..31 of the bottom tile are never used).
#pragma unroll
    for (int v = 0; v < 8; v++) hc[w][Mo + v][n] = ctop[v];
#pragma unroll
    for (int v = 0; v < 8; v++) {
      const int rowb = 16 + Mo + v;
      if (rowb <= 25) hc[w][rowb][n] = cbot[v];
    }
    // Vertical 11-tap conv (exact f32). Same-wave LDS deps -> HW DScnt order.
    float col[18];
#pragma unroll
    for (int r = 0; r < 18; r++) col[r] = hc[w][mbase + r][n];
#pragma unroll
    for (int j = 0; j < 8; j++) {
      float acc = 0.f;
#pragma unroll
      for (int d = 0; d < 11; d++) acc = fmaf(gw[d], col[j + d], acc);
      res[s][j] = acc;
    }
  }

  // SSIM map over this lane's 8 output pixels, masked to the 2038x2038 VALID
  // region, then wave-reduce and one f64 atomic per wave.
  float ss = 0.f;
#pragma unroll
  for (int j = 0; j < 8; j++) {
    const int ar = ty * 16 + mbase + j;
    const int ac = tx * 16 + n;
    if (ar < OUTD && ac < OUTD) {
      const float mu1 = res[0][j], mu2 = res[1][j];
      const float s11 = res[2][j] - mu1 * mu1;
      const float s22 = res[3][j] - mu2 * mu2;
      const float s12 = res[4][j] - mu1 * mu2;
      const float num = (2.f * mu1 * mu2 + C1v) * (2.f * s12 + C2v);
      const float den = (mu1 * mu1 + mu2 * mu2 + C1v) * (s11 + s22 + C2v);
      ss += num / den;
    }
  }
  ss = wred(ss);
  if (lane == 0) atomicAdd(ws + 24, (double)ss);
}

// ---------------------------------------------------------------------------
// Pass 3: finalize the 3 scalar outputs from the double accumulators.
// ---------------------------------------------------------------------------
__global__ void finalize_kernel(const double* __restrict__ ws,
                                float* __restrict__ out) {
  if (threadIdx.x == 0 && blockIdx.x == 0) {
    const double N = (double)IMG;
    const double l1 = ws[20] / (4.0 * N);
    const double grad = ws[21] / (4.0 * 2047.0 * 2048.0) +
                        ws[22] / (4.0 * 2048.0 * 2047.0);
    double energy = 0.0, dist = 0.0, psnr = 0.0;
    for (int b = 0; b < BATCH; b++) {
      const double pm = ws[b] / N, tm = ws[4 + b] / N;
      energy += (pm - tm) * (pm - tm);
      const double vps = (ws[8 + b] - N * pm * pm) / (N - 1.0);
      const double vts = (ws[12 + b] - N * tm * tm) / (N - 1.0);
      const double ps = sqrt(fmax(vps, 0.0));
      const double ts = sqrt(fmax(vts, 0.0));
      dist += (ps - ts) * (ps - ts);
      const double mse = ws[16 + b] / N;
      psnr += (mse == 0.0) ? 100.0 : 10.0 * log10(65025.0 / fmax(mse, 1e-12));
    }
    energy *= 0.25; dist *= 0.25; psnr *= 0.25;
    const double rng = ws[23] / (4.0 * N);
    const double phys = energy + 0.5 * dist + 0.1 * rng;
    double ssim = ws[24] / (4.0 * (double)OUTD * (double)OUTD);
    ssim = fmin(fmax(ssim, 0.0), 1.0);
    const double total = 1.0 * l1 + 0.15 * grad + 0.05 * phys +
                         0.1 * (1.0 - ssim);
    out[0] = (float)total;
    out[1] = (float)psnr;
    out[2] = (float)ssim;
  }
}

extern "C" void kernel_launch(void* const* d_in, const int* in_sizes, int n_in,
                              void* d_out, int out_size, void* d_ws,
                              size_t ws_size, hipStream_t stream) {
  (void)in_sizes; (void)n_in; (void)out_size; (void)ws_size;
  const float* pred = (const float*)d_in[0];
  const float* targ = (const float*)d_in[1];
  double* ws = (double*)d_ws;

  hipMemsetAsync(d_ws, 0, WS_DOUBLES * sizeof(double), stream);

  // Pass 1: 16,777,216 px / 1024 px-per-block = 16384 blocks
  stats_kernel<<<(BATCH * IMG) / 1024, 256, 0, stream>>>(pred, targ, ws);
  // Pass 2: 4*128*128 = 65536 tiles, 8 waves/block -> 8192 blocks
  ssim_kernel<<<(BATCH * 128 * 128) / 8, 256, 0, stream>>>(pred, targ, ws);
  // Pass 3
  finalize_kernel<<<1, 32, 0, stream>>>(ws, (float*)d_out);
}
